// XYZ_TimePiecewiseConstant_49813030699411
// MI455X (gfx1250) — matrix operand order repaired and checked by reading, not compiled, not run on hardware
//
#include <hip/hip_runtime.h>
#include <math.h>

typedef __attribute__((ext_vector_type(16))) _Float16 v16h;
typedef __attribute__((ext_vector_type(8)))  _Float16 v8h;
typedef __attribute__((ext_vector_type(2)))  _Float16 v2h;
typedef __attribute__((ext_vector_type(8)))  float    v8f;

#define N_LEVELS 16
#define N_PIECES 10
#define OUT_DIM  32      // N_LEVELS * N_FEAT, == WMMA K for f16
#define HIDDEN   64
#define INV_TEMP 0.01f

struct TablePtrs { const float* p[N_LEVELS]; };
struct ResArr    { int r[N_LEVELS]; };

// ---------------------------------------------------------------------------
// Kernel 1: multi-level trilinear gather -> emb (f16, row-major [B*10, 32])
// ---------------------------------------------------------------------------
__global__ void embed_kernel(const float* __restrict__ xyzt,
                             TablePtrs tabs, ResArr resa,
                             _Float16* __restrict__ embh, int B)
{
    int pt = blockIdx.x * blockDim.x + threadIdx.x;
    if (pt >= B) return;

    float x = xyzt[pt * 4 + 0];
    float y = xyzt[pt * 4 + 1];
    float z = xyzt[pt * 4 + 2];

    for (int lv = 0; lv < N_LEVELS; ++lv) {
        const int res = resa.r[lv];
        const float2* __restrict__ tb = (const float2*)tabs.p[lv];

        float px = x * (float)res, py = y * (float)res, pz = z * (float)res;
        int ix = (int)floorf(px), iy = (int)floorf(py), iz = (int)floorf(pz);
        float fx = px - (float)ix, fy = py - (float)iy, fz = pz - (float)iz;

        float2 acc[N_PIECES];
        #pragma unroll
        for (int j = 0; j < N_PIECES; ++j) { acc[j].x = 0.f; acc[j].y = 0.f; }

        #pragma unroll
        for (int c = 0; c < 8; ++c) {
            const int cx = (c >> 2) & 1, cy = (c >> 1) & 1, cz = c & 1;
            float w = (cx ? fx : 1.f - fx) * (cy ? fy : 1.f - fy) * (cz ? fz : 1.f - fz);
            int jx = min(ix + cx, res - 1);
            int jy = min(iy + cy, res - 1);
            int jz = min(iz + cz, res - 1);
            int flat = ((jx * res + jy) * res + jz) * N_PIECES;   // float2 entries
            const float2* e = tb + flat;                          // 80B contiguous
            #pragma unroll
            for (int j = 0; j < N_PIECES; ++j) {
                float2 v = e[j];
                acc[j].x = fmaf(w, v.x, acc[j].x);
                acc[j].y = fmaf(w, v.y, acc[j].y);
            }
        }
        #pragma unroll
        for (int j = 0; j < N_PIECES; ++j) {
            v2h h; h[0] = (_Float16)acc[j].x; h[1] = (_Float16)acc[j].y;
            *(v2h*)(embh + (size_t)(pt * N_PIECES + j) * OUT_DIM + 2 * lv) = h;
        }
    }
}

// ---------------------------------------------------------------------------
// Kernel 2: WMMA MLP. One wave -> 16 rows x 64 hidden via 4x
// v_wmma_f32_16x16x32_f16 (K=32 == OUT_DIM), ReLU, dot with W2,
// 16-lane butterfly reduce, sigmoid -> anchors[nrows]
// ---------------------------------------------------------------------------
__global__ void mlp_kernel(const _Float16* __restrict__ embh,
                           const float* __restrict__ W1,
                           const float* __restrict__ b1,
                           const float* __restrict__ W2,
                           const float* __restrict__ b2,
                           float* __restrict__ anchors, int nrows)
{
    const int lane = threadIdx.x & 31;
    const int c    = lane & 15;       // column within 16-wide tile / row id for A
    const int hiL  = lane >> 4;       // which K-half this lane holds
    const int wid  = (blockIdx.x * blockDim.x + threadIdx.x) >> 5;
    const int nw   = (gridDim.x * blockDim.x) >> 5;

    // Hoisted per-wave operands: B tiles of W1 (ISA 16-bit B layout:
    // lane holds column c; element k = K ((k>>3)*16 + hiL*8 + (k&7))), bias as C, W2 slice.
    v16h  bt[4];
    v8f   cb[4];
    float w2l[4];
    #pragma unroll
    for (int t = 0; t < 4; ++t) {
        #pragma unroll
        for (int k = 0; k < 16; ++k)
            bt[t][k] = (_Float16)W1[((k >> 3) * 16 + hiL * 8 + (k & 7)) * HIDDEN + t * 16 + c];
        float bv = b1[t * 16 + c];
        #pragma unroll
        for (int i = 0; i < 8; ++i) cb[t][i] = bv;
        w2l[t] = W2[t * 16 + c];
    }
    const float bias2 = b2[0];
    const int ntiles = nrows >> 4;

    for (int tile = wid; tile < ntiles; tile += nw) {
        // A operand, ISA 16-bit A layout: lane c holds row (tile*16+c);
        // elements 0..7 = K hiL*8 .. hiL*8+7, elements 8..15 = 16+hiL*8 ..
        const _Float16* rp = embh + (size_t)(tile * 16 + c) * OUT_DIM;
        v8h lo = *(const v8h*)(rp + hiL * 8);
        v8h hh = *(const v8h*)(rp + 16 + hiL * 8);
        v16h a;
        #pragma unroll
        for (int i = 0; i < 8; ++i) { a[i] = lo[i]; a[i + 8] = hh[i]; }

        float part[8];
        #pragma unroll
        for (int i = 0; i < 8; ++i) part[i] = 0.f;

        #pragma unroll
        for (int t = 0; t < 4; ++t) {
            v8f d = __builtin_amdgcn_wmma_f32_16x16x32_f16(
                false, a, false, bt[t], (short)0, cb[t], false, false);
            #pragma unroll
            for (int i = 0; i < 8; ++i)
                part[i] = fmaf(fmaxf(d[i], 0.f), w2l[t], part[i]);
        }

        // Reduce the 64 hidden cols: xor 1,2,4,8 stays inside each 16-lane
        // half (D layout: lanes 0-15 = rows 0-7, lanes 16-31 = rows 8-15).
        #pragma unroll
        for (int off = 1; off < 16; off <<= 1) {
            #pragma unroll
            for (int i = 0; i < 8; ++i)
                part[i] += __shfl_xor(part[i], off, 32);
        }

        if (c == 0) {
            #pragma unroll
            for (int i = 0; i < 8; ++i) {
                float v = part[i] + bias2;
                anchors[tile * 16 + hiL * 8 + i] = 1.0f / (1.0f + __expf(-v));
            }
        }
    }
}

// ---------------------------------------------------------------------------
// Kernel 3: softmax over 10 pieces + weighted sum of emb -> out (B, 32)
// One wave per point: lane = feature index.
// ---------------------------------------------------------------------------
__global__ void out_kernel(const float* __restrict__ xyzt,
                           const _Float16* __restrict__ embh,
                           const float* __restrict__ anchors,
                           float* __restrict__ out, int B)
{
    int gid = blockIdx.x * blockDim.x + threadIdx.x;
    int pt = gid >> 5;
    int f  = gid & 31;
    if (pt >= B) return;

    float t = xyzt[pt * 4 + 3];
    float a[N_PIECES];
    float m = -3.4e38f;
    #pragma unroll
    for (int p = 0; p < N_PIECES; ++p) {
        a[p] = -fabsf(t - anchors[pt * N_PIECES + p]) * INV_TEMP;
        m = fmaxf(m, a[p]);
    }
    float s = 0.f, acc = 0.f;
    #pragma unroll
    for (int p = 0; p < N_PIECES; ++p) {
        float w = __expf(a[p] - m);
        s += w;
        acc = fmaf(w, (float)embh[(size_t)(pt * N_PIECES + p) * OUT_DIM + f], acc);
    }
    out[pt * OUT_DIM + f] = acc / s;
}

// ---------------------------------------------------------------------------
// Host entry
// ---------------------------------------------------------------------------
extern "C" void kernel_launch(void* const* d_in, const int* in_sizes, int n_in,
                              void* d_out, int out_size, void* d_ws, size_t ws_size,
                              hipStream_t stream) {
    const float* xyzt = (const float*)d_in[0];
    TablePtrs tabs;
    ResArr    resa;
    for (int i = 0; i < N_LEVELS; ++i) {
        tabs.p[i] = (const float*)d_in[1 + i];
        // table has res^3 * N_PIECES * 2 floats -> res = cbrt(n / 20)
        double n = (double)in_sizes[1 + i];
        resa.r[i] = (int)(cbrt(n / 20.0) + 0.5);
    }
    const float* W1 = (const float*)d_in[17];
    const float* b1 = (const float*)d_in[18];
    const float* W2 = (const float*)d_in[19];
    const float* b2 = (const float*)d_in[20];

    const int B     = in_sizes[0] / 4;
    const int nrows = B * N_PIECES;

    _Float16* embh   = (_Float16*)d_ws;
    float*    anchors = (float*)((char*)d_ws + (size_t)nrows * OUT_DIM * sizeof(_Float16));
    float*    out    = (float*)d_out;

    embed_kernel<<<(B + 255) / 256, 256, 0, stream>>>(xyzt, tabs, resa, embh, B);
    mlp_kernel<<<1024, 256, 0, stream>>>(embh, W1, b1, W2, b2, anchors, nrows);
    out_kernel<<<(B * 32 + 255) / 256, 256, 0, stream>>>(xyzt, embh, anchors, out, B);
}
